// music_gen_53961969107307
// MI455X (gfx1250) — matrix operand order repaired and checked by reading, not compiled, not run on hardware
//
#include <hip/hip_runtime.h>
#include <cstdint>
#include <cstddef>

// ---------------------------------------------------------------------------
// CDNA5 (gfx1250) WMMA types
// ---------------------------------------------------------------------------
typedef __bf16 v16bf __attribute__((ext_vector_type(16)));
typedef float  v8f   __attribute__((ext_vector_type(8)));

union Frag16 {
  uint4          q[2];
  unsigned short u[16];
  v16bf          v;
};

__device__ __forceinline__ unsigned short f32_to_bf16(float f) {
  unsigned int x = __float_as_uint(f);
  unsigned int r = x + 0x7FFFu + ((x >> 16) & 1u);   // round-to-nearest-even
  return (unsigned short)(r >> 16);
}

__device__ __forceinline__ float sigmoidf_fast(float x) {
  return 1.0f / (1.0f + __expf(-x));
}

// ---------------------------------------------------------------------------
// Prep kernels
// ---------------------------------------------------------------------------

// x [16][127][78][82] f32 -> x1b bf16 [(b*78+n)][127][80]
__global__ void extract_x1_kernel(const float* __restrict__ x,
                                  unsigned short* __restrict__ x1b) {
  size_t idx = (size_t)blockIdx.x * blockDim.x + threadIdx.x;
  const size_t total = (size_t)1248 * 127 * 80;
  if (idx >= total) return;
  int c = (int)(idx % 80);
  size_t rt = idx / 80;
  int t = (int)(rt % 127);
  int r = (int)(rt / 127);
  int b = r / 78, n = r - b * 78;
  float v = x[(((size_t)b * 127 + t) * 78 + n) * 82 + c];
  x1b[idx] = f32_to_bf16(v);
}

// prevb bf16 [(b*127+t)][78] : n==0 ? 0 : x[b,t,n-1,81]
__global__ void extract_prev_kernel(const float* __restrict__ x,
                                    unsigned short* __restrict__ prevb) {
  int idx = blockIdx.x * blockDim.x + threadIdx.x;
  const int total = 2032 * 78;
  if (idx >= total) return;
  int n = idx % 78;
  int bt = idx / 78;   // b*127 + t
  float v = 0.0f;
  if (n > 0) v = x[(((size_t)bt) * 78 + (n - 1)) * 82 + 81];
  prevb[idx] = f32_to_bf16(v);
}

// Pack [Wk;Wr] (f32, layout [K][4H] Keras i,f,g,o) into bf16 [Kpad][4*Hc],
// columns zero-padded per gate to Hc; bias packed to f32 [4*Hc].
__global__ void pack_weights_kernel(const float* __restrict__ Wk,
                                    const float* __restrict__ Wr,
                                    const float* __restrict__ bias,
                                    int Kx, int Hr, int H, int Hc, int Kpad,
                                    unsigned short* __restrict__ Wc,
                                    float* __restrict__ bc) {
  int idx = blockIdx.x * blockDim.x + threadIdx.x;
  int total = Kpad * 4 * Hc;
  if (idx >= total) return;
  int k  = idx / (4 * Hc);
  int c4 = idx - k * 4 * Hc;
  int g  = c4 / Hc;
  int j  = c4 - g * Hc;
  float v = 0.0f;
  if (j < H) {
    int src = g * H + j;
    if (k < Kx)            v = Wk[(size_t)k * (4 * H) + src];
    else if (k < Kx + Hr)  v = Wr[(size_t)(k - Kx) * (4 * H) + src];
    if (k == 0) bc[c4] = bias[src];
  } else if (k == 0) {
    bc[c4] = 0.0f;
  }
  Wc[idx] = f32_to_bf16(v);
}

// ---------------------------------------------------------------------------
// Per-layer A-operand element fetch: concat(x_t, h_prev)[row][k] in bf16 bits
// ---------------------------------------------------------------------------
__device__ __forceinline__ unsigned short fetchA(
    int layer, int row, int k, int S, int t,
    const unsigned short* __restrict__ xsrc,
    const unsigned short* __restrict__ prevb,
    const unsigned short* __restrict__ hstate) {
  unsigned short val = 0;
  if (layer == 1) {
    if (k < 80)       val = xsrc[((size_t)row * S + t) * 80 + k];
    else if (k < 380) val = hstate[(size_t)row * 300 + (k - 80)];
  } else if (layer == 2) {
    if (k < 300)      val = xsrc[((size_t)row * S + t) * 300 + k];
    else if (k < 600) val = hstate[(size_t)row * 300 + (k - 300)];
  } else if (layer == 3) {
    if (k < 300) {
      int b = row / 127, tt = row - b * 127;   // row = b*127 + tt
      val = xsrc[(((size_t)(b * 78 + t)) * 127 + tt) * 300 + k];
    } else if (k == 300) val = prevb[(size_t)row * 78 + t];
    else if (k < 401)    val = hstate[(size_t)row * 100 + (k - 301)];
  } else {  // layer 4
    if (k < 100)       val = xsrc[((size_t)row * S + t) * 100 + k];
    else if (k == 100) val = prevb[(size_t)row * 78 + t];
    else if (k < 151)  val = hstate[(size_t)row * 50 + (k - 101)];
  }
  return val;
}

// ---------------------------------------------------------------------------
// Fused LSTM step: each wave computes a 16-row x 16-unit tile, accumulating
// four WMMA chains (gates i,f,g,o) over K = concat(x_t, h_prev), then does
// the c/h update in-register.  A tile double-buffered in LDS: next chunk's
// global loads + ds_stores overlap the current chunk's WMMAs; one barrier
// per chunk.
// ---------------------------------------------------------------------------
__global__ __launch_bounds__(256) void lstm_step_kernel(
    int layer, int t, int S, int H, int Hc, int Kpad,
    const unsigned short* __restrict__ Wc, const float* __restrict__ bc,
    const unsigned short* __restrict__ xsrc,
    const unsigned short* __restrict__ prevb,
    unsigned short* __restrict__ hstate, float* __restrict__ cstate,
    unsigned short* __restrict__ hseq) {
  __shared__ __align__(16) unsigned short As[2][16 * 40];  // double-buffered

  const int tid  = threadIdx.x;
  const int wave = tid >> 5;
  const int lane = tid & 31;
  const int rowBase = blockIdx.y * 16;
  const int nTiles  = (H + 15) >> 4;
  const int hTile   = blockIdx.x * 8 + wave;
  const bool active = hTile < nTiles;
  const int ld      = 4 * Hc;
  const int nChunks = Kpad >> 5;

  // staging indices for this thread (2 elements of the 16x32 tile)
  const int sm0 = tid >> 4;            // 0..15
  const int sk0 = (tid & 15) * 2;      // 0,2,..,30

  v8f acc[4];
#pragma unroll
  for (int g = 0; g < 4; ++g) acc[g] = v8f{0, 0, 0, 0, 0, 0, 0, 0};

  // prologue: stage chunk 0 into buffer 0
  {
    int row = rowBase + sm0;
    As[0][sm0 * 40 + sk0]     = fetchA(layer, row, sk0,     S, t, xsrc, prevb, hstate);
    As[0][sm0 * 40 + sk0 + 1] = fetchA(layer, row, sk0 + 1, S, t, xsrc, prevb, hstate);
  }

  for (int ci = 0; ci < nChunks; ++ci) {
    const int cur = ci & 1;
    const int kc  = ci << 5;
    __syncthreads();   // chunk ci staged; previous reads of buf `cur` done

    // stage chunk ci+1 into the other buffer (overlaps WMMAs below)
    if (ci + 1 < nChunks) {
      int kn  = kc + 32;
      int row = rowBase + sm0;
      As[cur ^ 1][sm0 * 40 + sk0]     = fetchA(layer, row, kn + sk0,     S, t, xsrc, prevb, hstate);
      As[cur ^ 1][sm0 * 40 + sk0 + 1] = fetchA(layer, row, kn + sk0 + 1, S, t, xsrc, prevb, hstate);
    }

    if (active) {
      const int m  = lane & 15;
      const int hi = lane >> 4;
      // A fragment: e<8 -> K = 8*hi+e ; e>=8 -> K = 16 + 8*hi + (e-8)
      Frag16 a;
      a.q[0] = *reinterpret_cast<const uint4*>(&As[cur][m * 40 + 8 * hi]);
      a.q[1] = *reinterpret_cast<const uint4*>(&As[cur][m * 40 + 16 + 8 * hi]);
      const size_t krow = (size_t)(kc + lane) * ld;
      // prefetch next chunk's B rows into WGP$ (global_prefetch_b8)
      if (kc + 32 < Kpad) {
        const unsigned short* pf = Wc + (size_t)(kc + 32 + lane) * ld + hTile * 16;
        __builtin_prefetch(pf, 0, 1);
        __builtin_prefetch(pf + 2 * Hc, 0, 1);
      }
#pragma unroll
      for (int g = 0; g < 4; ++g) {
        const unsigned short* pb = Wc + g * Hc + hTile * 16 + (lane & 15);
        Frag16 b;
#pragma unroll
        for (int e = 0; e < 16; ++e) {
          int k = kc + (e < 8 ? 8 * hi + e : 16 + 8 * hi + (e - 8));
          b.u[e] = pb[(size_t)k * ld];
        }
        acc[g] = __builtin_amdgcn_wmma_f32_16x16x32_bf16(
            false, a.v, false, b.v, (short)0, acc[g], false, false);
      }
    }
  }

  // --- gate nonlinearity + state update, in-register ---
  if (active) {
    const int n  = lane & 15;
    const int hi = lane >> 4;
    const int j  = hTile * 16 + n;
    if (j < H) {
      float bi = bc[0 * Hc + j];
      float bf = bc[1 * Hc + j];
      float bg = bc[2 * Hc + j];
      float bo = bc[3 * Hc + j];
#pragma unroll
      for (int i = 0; i < 8; ++i) {
        int row = rowBase + i + 8 * hi;
        float zi = acc[0][i] + bi;
        float zf = acc[1][i] + bf;
        float zg = acc[2][i] + bg;
        float zo = acc[3][i] + bo;
        float si = sigmoidf_fast(zi);
        float sf = sigmoidf_fast(zf);
        float so = sigmoidf_fast(zo);
        float tg = tanhf(zg);
        size_t cs = (size_t)row * H + j;
        float c = sf * cstate[cs] + si * tg;
        cstate[cs] = c;
        float h = so * tanhf(c);
        unsigned short hb = f32_to_bf16(h);
        hstate[cs] = hb;
        hseq[((size_t)row * S + t) * H + j] = hb;
      }
    }
  }
}

// ---------------------------------------------------------------------------
// Dense head: out[16][156] = sigmoid(h4[16][495300] @ Wd + bd).
// 10 column tiles; 8 waves split-K per block, LDS reduction.  Wd converted
// f32 -> bf16 on the fly.
// ---------------------------------------------------------------------------
__global__ __launch_bounds__(256) void dense_kernel(
    const unsigned short* __restrict__ h4, const float* __restrict__ Wd,
    const float* __restrict__ bd, float* __restrict__ out) {
  const int K = 495300;
  const int tile = blockIdx.x;          // 0..9
  const int col0 = tile * 16;
  const int wave = threadIdx.x >> 5;
  const int lane = threadIdx.x & 31;
  const int m  = lane & 15;
  const int hi = lane >> 4;

  __shared__ float red[8 * 256];

  v8f acc = v8f{0, 0, 0, 0, 0, 0, 0, 0};
  const int nChunks = (K + 31) / 32;    // 15479 (last chunk partially masked)

  for (int ci = wave; ci < nChunks; ci += 8) {
    int kc = ci * 32;
    Frag16 a;
#pragma unroll
    for (int e = 0; e < 16; ++e) {
      int k = kc + (e < 8 ? 8 * hi + e : 16 + 8 * hi + (e - 8));
      a.u[e] = (k < K) ? h4[(size_t)m * K + k] : (unsigned short)0;
    }
    Frag16 b;
    int col = col0 + m;
#pragma unroll
    for (int e = 0; e < 16; ++e) {
      int kb = kc + (e < 8 ? 8 * hi + e : 16 + 8 * hi + (e - 8));
      float w = (kb < K && col < 156) ? Wd[(size_t)kb * 156 + col] : 0.0f;
      b.u[e] = f32_to_bf16(w);
    }
    acc = __builtin_amdgcn_wmma_f32_16x16x32_bf16(
        false, a.v, false, b.v, (short)0, acc, false, false);
  }

#pragma unroll
  for (int i = 0; i < 8; ++i) red[wave * 256 + lane * 8 + i] = acc[i];
  __syncthreads();

  if (wave == 0) {
#pragma unroll
    for (int i = 0; i < 8; ++i) {
      float s = 0.0f;
      for (int w = 0; w < 8; ++w) s += red[w * 256 + lane * 8 + i];
      int row = i + 8 * hi;             // batch index 0..15
      int col = col0 + (lane & 15);
      if (col < 156) {
        float z = s + bd[col];
        out[row * 156 + col] = sigmoidf_fast(z);
      }
    }
  }
}

// ---------------------------------------------------------------------------
// Host side
// ---------------------------------------------------------------------------
extern "C" void kernel_launch(void* const* d_in, const int* in_sizes, int n_in,
                              void* d_out, int out_size, void* d_ws,
                              size_t ws_size, hipStream_t stream) {
  const float* x   = (const float*)d_in[0];
  const float* W1k = (const float*)d_in[1];
  const float* W1r = (const float*)d_in[2];
  const float* b1  = (const float*)d_in[3];
  const float* W2k = (const float*)d_in[4];
  const float* W2r = (const float*)d_in[5];
  const float* b2  = (const float*)d_in[6];
  const float* W3k = (const float*)d_in[7];
  const float* W3r = (const float*)d_in[8];
  const float* b3  = (const float*)d_in[9];
  const float* W4k = (const float*)d_in[10];
  const float* W4r = (const float*)d_in[11];
  const float* b4  = (const float*)d_in[12];
  const float* Wd  = (const float*)d_in[13];
  const float* bd  = (const float*)d_in[14];
  float* out = (float*)d_out;

  char* ws = (char*)d_ws;
  size_t off = 0;
  auto alloc = [&](size_t bytes) -> char* {
    char* p = ws + off;
    off += (bytes + 255) & ~(size_t)255;
    return p;
  };

  // bf16 activation buffers
  unsigned short* x1b   = (unsigned short*)alloc((size_t)1248 * 127 * 80 * 2);
  unsigned short* prevb = (unsigned short*)alloc((size_t)2032 * 78 * 2);
  unsigned short* h1seq = (unsigned short*)alloc((size_t)1248 * 127 * 300 * 2);
  unsigned short* h2seq = (unsigned short*)alloc((size_t)1248 * 127 * 300 * 2);
  unsigned short* h3seq = (unsigned short*)alloc((size_t)2032 * 78 * 100 * 2);
  unsigned short* h4seq = (unsigned short*)alloc((size_t)2032 * 78 * 50 * 2);
  // recurrent state (reused across layers)
  unsigned short* hstate = (unsigned short*)alloc((size_t)1248 * 300 * 2);
  float*          cstate = (float*)alloc((size_t)1248 * 300 * 4);
  // packed weights  (Kpad x 4*Hc bf16) + packed bias (f32)
  unsigned short* Wc1 = (unsigned short*)alloc((size_t)384 * 1216 * 2);
  float*          bc1 = (float*)alloc((size_t)1216 * 4);
  unsigned short* Wc2 = (unsigned short*)alloc((size_t)608 * 1216 * 2);
  float*          bc2 = (float*)alloc((size_t)1216 * 4);
  unsigned short* Wc3 = (unsigned short*)alloc((size_t)416 * 448 * 2);
  float*          bc3 = (float*)alloc((size_t)448 * 4);
  unsigned short* Wc4 = (unsigned short*)alloc((size_t)160 * 256 * 2);
  float*          bc4 = (float*)alloc((size_t)256 * 4);

  // ---- prep ----
  {
    size_t total = (size_t)1248 * 127 * 80;
    extract_x1_kernel<<<(unsigned)((total + 255) / 256), 256, 0, stream>>>(x, x1b);
  }
  extract_prev_kernel<<<(2032 * 78 + 255) / 256, 256, 0, stream>>>(x, prevb);

  pack_weights_kernel<<<(384 * 1216 + 255) / 256, 256, 0, stream>>>(
      W1k, W1r, b1, 80, 300, 300, 304, 384, Wc1, bc1);
  pack_weights_kernel<<<(608 * 1216 + 255) / 256, 256, 0, stream>>>(
      W2k, W2r, b2, 300, 300, 300, 304, 608, Wc2, bc2);
  pack_weights_kernel<<<(416 * 448 + 255) / 256, 256, 0, stream>>>(
      W3k, W3r, b3, 301, 100, 100, 112, 416, Wc3, bc3);
  pack_weights_kernel<<<(160 * 256 + 255) / 256, 256, 0, stream>>>(
      W4k, W4r, b4, 101, 50, 50, 64, 160, Wc4, bc4);

  // ---- LSTM layer 1: rows=1248, S=127, H=300 ----
  hipMemsetAsync(hstate, 0, (size_t)1248 * 300 * 2, stream);
  hipMemsetAsync(cstate, 0, (size_t)1248 * 300 * 4, stream);
  for (int t = 0; t < 127; ++t)
    lstm_step_kernel<<<dim3(3, 78), 256, 0, stream>>>(
        1, t, 127, 300, 304, 384, Wc1, bc1, x1b, nullptr, hstate, cstate, h1seq);

  // ---- LSTM layer 2 ----
  hipMemsetAsync(hstate, 0, (size_t)1248 * 300 * 2, stream);
  hipMemsetAsync(cstate, 0, (size_t)1248 * 300 * 4, stream);
  for (int t = 0; t < 127; ++t)
    lstm_step_kernel<<<dim3(3, 78), 256, 0, stream>>>(
        2, t, 127, 300, 304, 608, Wc2, bc2, h1seq, nullptr, hstate, cstate, h2seq);

  // ---- LSTM layer 3: rows=2032, S=78, H=100 ----
  hipMemsetAsync(hstate, 0, (size_t)2032 * 100 * 2, stream);
  hipMemsetAsync(cstate, 0, (size_t)2032 * 100 * 4, stream);
  for (int t = 0; t < 78; ++t)
    lstm_step_kernel<<<dim3(1, 127), 256, 0, stream>>>(
        3, t, 78, 100, 112, 416, Wc3, bc3, h2seq, prevb, hstate, cstate, h3seq);

  // ---- LSTM layer 4: rows=2032, S=78, H=50 ----
  hipMemsetAsync(hstate, 0, (size_t)2032 * 50 * 2, stream);
  hipMemsetAsync(cstate, 0, (size_t)2032 * 50 * 4, stream);
  for (int t = 0; t < 78; ++t)
    lstm_step_kernel<<<dim3(1, 127), 256, 0, stream>>>(
        4, t, 78, 50, 64, 160, Wc4, bc4, h3seq, prevb, hstate, cstate, h4seq);

  // ---- dense + sigmoid ----
  dense_kernel<<<10, 256, 0, stream>>>(h4seq, Wd, bd, out);
}
